// grafiti_78795470012896
// MI455X (gfx1250) — compile-verified, matrix operand order checked by reading, not run on hardware
//
#include <hip/hip_runtime.h>
#include <hip/hip_bf16.h>

// ---------------- problem constants ----------------
#define B_  8
#define T_  256
#define C_  32
#define D_  64
#define H_  2
#define NL_ 3
#define L_  8192   // T_*C_

typedef __attribute__((ext_vector_type(16))) _Float16 v16h;
typedef __attribute__((ext_vector_type(8)))  float    v8f;

// K index mapping for 16-bit A/B WMMA fragments (ISA 7.12.2):
// lanes 0-15 hold K = j (j<8) and 16+(j-8) (j>=8); lanes 16-31 add +8.
__device__ __forceinline__ int kmap(int hf, int j) {
    return (hf << 3) + j + ((j >= 8) << 3);
}

// ---------------------------------------------------------------------------
// 1. Stable partition per batch: valid entries first, row-major order.
//    Also emits mk, T_inds, C_inds, U2 and per-channel/time valid counts.
// ---------------------------------------------------------------------------
__global__ __launch_bounds__(256) void build_order_k(
    const float* __restrict__ MX, const float* __restrict__ MY,
    const float* __restrict__ X,
    int* __restrict__ order, int* __restrict__ nvalid,
    float* __restrict__ mk, int* __restrict__ Tind, int* __restrict__ Cind,
    float* __restrict__ U2, int* __restrict__ countC, int* __restrict__ countT)
{
    int b = blockIdx.x, tid = threadIdx.x;
    __shared__ int sc[256];
    __shared__ int basev, basei;
    const float* mx = MX + (size_t)b * L_;
    const float* my = MY + (size_t)b * L_;
    const float* xx = X  + (size_t)b * L_;

    for (int i = tid; i < C_; i += 256) countC[b * C_ + i] = 0;
    for (int i = tid; i < T_; i += 256) countT[b * T_ + i] = 0;

    // pass 1: total valid count
    int cnt = 0;
    for (int l = tid; l < L_; l += 256) cnt += ((mx[l] + my[l]) > 0.f);
    sc[tid] = cnt; __syncthreads();
    for (int s = 128; s > 0; s >>= 1) { if (tid < s) sc[tid] += sc[tid + s]; __syncthreads(); }
    if (tid == 0) { nvalid[b] = sc[0]; basev = 0; basei = sc[0]; }
    __syncthreads();

    // pass 2: stable partition, 256 elements per chunk
    for (int chunk = 0; chunk < L_ / 256; ++chunk) {
        int l = chunk * 256 + tid;
        float m = mx[l] + my[l];
        int v = (m > 0.f) ? 1 : 0;
        sc[tid] = v; __syncthreads();
        for (int s = 1; s < 256; s <<= 1) {
            int t = (tid >= s) ? sc[tid - s] : 0; __syncthreads();
            sc[tid] += t; __syncthreads();
        }
        int excl = sc[tid] - v;
        int pos = v ? (basev + excl) : (basei + (tid - excl));
        size_t dp = (size_t)b * L_ + pos;
        order[dp] = l;
        mk[dp]    = m;                       // invalid => m == 0 exactly
        Tind[dp]  = v ? (l / C_) : 0;
        Cind[dp]  = v ? (l % C_) : 0;
        float tg  = v ? my[l] : 0.f;
        U2[dp * 2]     = v ? xx[l] : 0.f;
        U2[dp * 2 + 1] = 1.f - m + tg;
        if (v) {
            atomicAdd(&countC[b * C_ + (l % C_)], 1);
            atomicAdd(&countT[b * T_ + (l / C_)], 1);
        }
        __syncthreads();
        if (tid == 0) { basev += sc[255]; basei += 256 - sc[255]; }
        __syncthreads();
    }
}

__global__ void make_cmtm_k(const int* __restrict__ countC, const int* __restrict__ countT,
                            float* __restrict__ CmF, float* __restrict__ TmF)
{
    int i = blockIdx.x * 256 + threadIdx.x;
    if (i < B_ * C_) CmF[i] = (countC[i] > 0) ? 1.f : 0.f;
    if (i < B_ * T_) TmF[i] = (countT[i] > 0) ? 1.f : 0.f;
}

// ---------------------------------------------------------------------------
// 2. Feature initialization
// ---------------------------------------------------------------------------
__global__ void edge_embed_k(const float* __restrict__ U2, const float* __restrict__ mk,
                             const float* __restrict__ ew, const float* __restrict__ eb,
                             float* __restrict__ U)
{
    size_t idx = (size_t)blockIdx.x * 256 + threadIdx.x;
    if (idx >= (size_t)B_ * L_ * D_) return;
    size_t row = idx >> 6; int d = idx & 63;
    float v = fmaf(U2[row * 2], ew[d], fmaf(U2[row * 2 + 1], ew[D_ + d], eb[d]));
    U[idx] = fmaxf(v, 0.f) * mk[row];
}

__global__ void tf_init_k(const float* __restrict__ TX, const float* __restrict__ tw,
                          const float* __restrict__ tb, float* __restrict__ Tf)
{
    size_t idx = (size_t)blockIdx.x * 256 + threadIdx.x;
    if (idx >= (size_t)B_ * T_ * D_) return;
    size_t bt = idx >> 6; int d = idx & 63;
    Tf[idx] = __sinf(fmaf(TX[bt], tw[d], tb[d]));
}

__global__ void cf_init_k(const float* __restrict__ cw, const float* __restrict__ cb,
                          float* __restrict__ Cf)
{
    size_t idx = (size_t)blockIdx.x * 256 + threadIdx.x;
    if (idx >= (size_t)B_ * C_ * D_) return;
    int c = (int)((idx >> 6) % C_); int d = idx & 63;
    Cf[idx] = fmaxf(cw[c * D_ + d] + cb[d], 0.f);
}

// ---------------------------------------------------------------------------
// 3. Helpers: weight transpose+f16 convert, f32->f16 pack, gather, scatter
// ---------------------------------------------------------------------------
__global__ void wt_transpose_k(const float* __restrict__ W, _Float16* __restrict__ Wt,
                               int Kd, int Nd)
{
    int idx = blockIdx.x * 256 + threadIdx.x;
    if (idx >= Kd * Nd) return;
    int k = idx / Nd, n = idx % Nd;
    Wt[n * Kd + k] = (_Float16)W[idx];
}

__global__ void pack_f16_k(const float* __restrict__ src, _Float16* __restrict__ dst, int n)
{
    int idx = blockIdx.x * 256 + threadIdx.x;
    if (idx < n) dst[idx] = (_Float16)src[idx];
}

__global__ __launch_bounds__(64) void gather_k(
    const float* __restrict__ Tf, const float* __restrict__ Cf,
    const float* __restrict__ U,
    const int* __restrict__ Tind, const int* __restrict__ Cind,
    _Float16* __restrict__ kvinC, _Float16* __restrict__ kvinT,
    _Float16* __restrict__ eg)
{
    size_t row = blockIdx.x; int d = threadIdx.x;
    int b = (int)(row >> 13);                       // row / L_
    int ti = Tind[row], ci = Cind[row];
    float kt = Tf[((size_t)b * T_ + ti) * D_ + d];
    float kc = Cf[((size_t)b * C_ + ci) * D_ + d];
    float u  = U[row * D_ + d];
    kvinC[row * 128 + d]       = (_Float16)kt;      // C-MAB keys: [k_t | U]
    kvinC[row * 128 + 64 + d]  = (_Float16)u;
    kvinT[row * 128 + d]       = (_Float16)kc;      // T-MAB keys: [k_c | U]
    kvinT[row * 128 + 64 + d]  = (_Float16)u;
    eg[row * 192 + d]          = (_Float16)u;       // edgenn: [U | k_t | k_c]
    eg[row * 192 + 64 + d]     = (_Float16)kt;
    eg[row * 192 + 128 + d]    = (_Float16)kc;
}

__global__ __launch_bounds__(64) void scatter_out_k(
    const float* __restrict__ U, const int* __restrict__ order,
    const int* __restrict__ nvalid, float* __restrict__ out)
{
    size_t p = blockIdx.x; int d = threadIdx.x;
    int b = (int)(p >> 13);
    int o = order[p];
    float v = ((int)(p & (L_ - 1)) < nvalid[b]) ? U[p * D_ + d] : 0.f;
    out[((size_t)b * L_ + o) * D_ + d] = v;
}

// ---------------------------------------------------------------------------
// 4. Generic fused WMMA GEMM:  Y = post( X[M,K](f16) @ W[K,64] + bias )
//    Wt is pre-transposed f16 [64, K].  post = (+resid) -> relu? -> *rowscale.
//    One block = 16 M-rows; 4 waves cover the 4 N-subtiles of N=64.
// ---------------------------------------------------------------------------
__global__ __launch_bounds__(128) void gemm_wmma_k(
    const _Float16* __restrict__ X, const _Float16* __restrict__ Wt,
    const float* __restrict__ bias, const float* __restrict__ resid,
    const float* __restrict__ rowscale,
    float* __restrict__ Yf, _Float16* __restrict__ Yh,
    int M, int K, int relu)
{
    const int wave = threadIdx.x >> 5;
    const int lane = threadIdx.x & 31;
    const int hf = lane >> 4, ln = lane & 15;
    const int n0 = wave * 16;
    const int m0 = blockIdx.x * 16;

    v8f acc = {0.f, 0.f, 0.f, 0.f, 0.f, 0.f, 0.f, 0.f};
    const _Float16* xr = X  + (size_t)(m0 + ln) * K;
    const _Float16* wr = Wt + (size_t)(n0 + ln) * K;
    for (int kk = 0; kk < K; kk += 32) {
        v16h a, bf;
#pragma unroll
        for (int j = 0; j < 16; ++j) {
            int kx = kmap(hf, j);
            a[j]  = xr[kk + kx];
            bf[j] = wr[kk + kx];
        }
        acc = __builtin_amdgcn_wmma_f32_16x16x32_f16(false, a, false, bf,
                                                     (short)0, acc, false, false);
    }
#pragma unroll
    for (int r = 0; r < 8; ++r) {
        int m = m0 + r + (hf << 3);
        int n = n0 + ln;
        float v = acc[r] + bias[n];
        if (resid)    v += resid[(size_t)m * D_ + n];
        if (relu)     v = fmaxf(v, 0.f);
        if (rowscale) v *= rowscale[m];
        if (Yf) Yf[(size_t)m * D_ + n] = v;
        if (Yh) Yh[(size_t)m * D_ + n] = (_Float16)v;
    }
}

// ---------------------------------------------------------------------------
// 5. Fused flash attention (one wave per (b, h, 16-query tile)).
//    QK^T via WMMA, masked online softmax, P.V via WMMA (P staged in LDS to
//    re-pack C-layout -> A-fragment layout).  Matches reference -1e8 masking.
// ---------------------------------------------------------------------------
__global__ __launch_bounds__(32) void attn_k(
    const _Float16* __restrict__ qh, const _Float16* __restrict__ kh,
    const _Float16* __restrict__ vh, const int* __restrict__ inds,
    const float* __restrict__ mk, _Float16* __restrict__ out, int S)
{
    const int qt_per = S >> 4;
    const int blk = blockIdx.x;
    const int b  = blk / (H_ * qt_per);
    const int rm = blk % (H_ * qt_per);
    const int h  = rm / qt_per;
    const int qt = rm % qt_per;
    const int lane = threadIdx.x;
    const int hf = lane >> 4, ln = lane & 15;

    __shared__ _Float16 pt[2][16][17];

    v16h qa;
    {
        const _Float16* qr = qh + ((size_t)(b * S + qt * 16 + ln)) * D_ + h * 32;
#pragma unroll
        for (int j = 0; j < 16; ++j) qa[j] = qr[kmap(hf, j)];
    }

    float mrow[8], lrow[8], pr[2][8];
#pragma unroll
    for (int r = 0; r < 8; ++r) { mrow[r] = -3.0e38f; lrow[r] = 0.f; }
    v8f o0 = {0.f,0.f,0.f,0.f,0.f,0.f,0.f,0.f};
    v8f o1 = {0.f,0.f,0.f,0.f,0.f,0.f,0.f,0.f};
    const float scale = 0.17677669529663687f;   // 1/sqrt(dk), dk = 32

    for (int kb2 = 0; kb2 < L_ / 32; ++kb2) {
#pragma unroll
        for (int sub = 0; sub < 2; ++sub) {
            int keyb = kb2 * 32 + sub * 16;
            v16h kf;
            const _Float16* kr = kh + ((size_t)(b * L_ + keyb + ln)) * D_ + h * 32;
#pragma unroll
            for (int j = 0; j < 16; ++j) kf[j] = kr[kmap(hf, j)];
            v8f s = {0.f,0.f,0.f,0.f,0.f,0.f,0.f,0.f};
            s = __builtin_amdgcn_wmma_f32_16x16x32_f16(false, qa, false, kf,
                                                       (short)0, s, false, false);
            int key = keyb + ln;
            int ki  = inds[(size_t)b * L_ + key];
            int kval = (mk[(size_t)b * L_ + key] > 0.f) ? 1 : 0;
#pragma unroll
            for (int r = 0; r < 8; ++r) {
                int q = qt * 16 + r + (hf << 3);
                float sv = (kval && ki == q) ? s[r] * scale : -1.0e8f;
                float tm = sv;
                tm = fmaxf(tm, __shfl_xor(tm, 1, 32));
                tm = fmaxf(tm, __shfl_xor(tm, 2, 32));
                tm = fmaxf(tm, __shfl_xor(tm, 4, 32));
                tm = fmaxf(tm, __shfl_xor(tm, 8, 32));
                float mn   = fmaxf(mrow[r], tm);
                float corr = __expf(mrow[r] - mn);
                float p    = __expf(sv - mn);
                float rs = p;
                rs += __shfl_xor(rs, 1, 32);
                rs += __shfl_xor(rs, 2, 32);
                rs += __shfl_xor(rs, 4, 32);
                rs += __shfl_xor(rs, 8, 32);
                lrow[r] = lrow[r] * corr + rs;
                mrow[r] = mn;
                o0[r] *= corr; o1[r] *= corr;
                if (sub == 1) pr[0][r] *= corr;   // rescale buffered sub-tile
                pr[sub][r] = p;
            }
        }
        // stage P (C-layout) into LDS, reload as A fragment
#pragma unroll
        for (int r = 0; r < 8; ++r) {
            pt[0][r + (hf << 3)][ln] = (_Float16)pr[0][r];
            pt[1][r + (hf << 3)][ln] = (_Float16)pr[1][r];
        }
        __syncthreads();
        v16h pa, v0, v1;
#pragma unroll
        for (int j = 0; j < 16; ++j) {
            int kx = kmap(hf, j);                 // key index in [0,32)
            pa[j] = pt[kx >> 4][ln][kx & 15];
            const _Float16* vr = vh + ((size_t)(b * L_ + kb2 * 32 + kx)) * D_ + h * 32;
            v0[j] = vr[ln];
            v1[j] = vr[16 + ln];
        }
        o0 = __builtin_amdgcn_wmma_f32_16x16x32_f16(false, pa, false, v0,
                                                    (short)0, o0, false, false);
        o1 = __builtin_amdgcn_wmma_f32_16x16x32_f16(false, pa, false, v1,
                                                    (short)0, o1, false, false);
        __syncthreads();
    }
#pragma unroll
    for (int r = 0; r < 8; ++r) {
        int q = qt * 16 + r + (hf << 3);
        float invl = 1.0f / lrow[r];
        size_t o = ((size_t)(b * S + q)) * D_ + h * 32;
        out[o + ln]      = (_Float16)(o0[r] * invl);
        out[o + 16 + ln] = (_Float16)(o1[r] * invl);
    }
}

// ---------------------------------------------------------------------------
// Host orchestration
// ---------------------------------------------------------------------------
extern "C" void kernel_launch(void* const* d_in, const int* in_sizes, int n_in,
                              void* d_out, int out_size, void* d_ws, size_t ws_size,
                              hipStream_t stream)
{
    (void)in_sizes; (void)n_in; (void)out_size; (void)ws_size;
    const float* TX = (const float*)d_in[0];
    const float* X  = (const float*)d_in[1];
    const float* MX = (const float*)d_in[2];
    const float* MY = (const float*)d_in[3];
    const float* edge_w = (const float*)d_in[4];
    const float* edge_b = (const float*)d_in[5];
    const float* chan_w = (const float*)d_in[6];
    const float* chan_b = (const float*)d_in[7];
    const float* time_w = (const float*)d_in[8];
    const float* time_b = (const float*)d_in[9];
    const float* qw = (const float*)d_in[10];
    const float* qb = (const float*)d_in[11];
    const float* kw = (const float*)d_in[12];
    const float* kb = (const float*)d_in[13];
    const float* vw = (const float*)d_in[14];
    const float* vb = (const float*)d_in[15];
    const float* ow = (const float*)d_in[16];
    const float* ob = (const float*)d_in[17];
    const float* ew = (const float*)d_in[18];
    const float* eb = (const float*)d_in[19];
    float* out = (float*)d_out;

    char* ws = (char*)d_ws;
    size_t cur = 0;
    auto alloc = [&](size_t bytes) -> void* {
        void* p = ws + cur;
        cur = (cur + bytes + 255) & ~(size_t)255;
        return p;
    };
    const size_t BL = (size_t)B_ * L_;

    int*   order  = (int*)  alloc(BL * 4);
    int*   nvalid = (int*)  alloc(B_ * 4);
    int*   countC = (int*)  alloc(B_ * C_ * 4);
    int*   countT = (int*)  alloc(B_ * T_ * 4);
    float* CmF    = (float*)alloc(B_ * C_ * 4);
    float* TmF    = (float*)alloc(B_ * T_ * 4);
    float* mkb    = (float*)alloc(BL * 4);
    int*   Tind   = (int*)  alloc(BL * 4);
    int*   Cind   = (int*)  alloc(BL * 4);
    float* U2     = (float*)alloc(BL * 2 * 4);
    float* Ubuf[2]  = { (float*)alloc(BL * D_ * 4), (float*)alloc(BL * D_ * 4) };
    float* Tfb[2]   = { (float*)alloc((size_t)B_ * T_ * D_ * 4), (float*)alloc((size_t)B_ * T_ * D_ * 4) };
    float* Cfb[2]   = { (float*)alloc((size_t)B_ * C_ * D_ * 4), (float*)alloc((size_t)B_ * C_ * D_ * 4) };
    _Float16* kvinC = (_Float16*)alloc(BL * 128 * 2);
    _Float16* kvinT = (_Float16*)alloc(BL * 128 * 2);
    _Float16* eg    = (_Float16*)alloc(BL * 192 * 2);
    _Float16* khC   = (_Float16*)alloc(BL * D_ * 2);
    _Float16* vhC   = (_Float16*)alloc(BL * D_ * 2);
    _Float16* khT   = (_Float16*)alloc(BL * D_ * 2);
    _Float16* vhT   = (_Float16*)alloc(BL * D_ * 2);
    _Float16* qinC  = (_Float16*)alloc((size_t)B_ * C_ * D_ * 2);
    _Float16* qinT  = (_Float16*)alloc((size_t)B_ * T_ * D_ * 2);
    _Float16* qhC   = (_Float16*)alloc((size_t)B_ * C_ * D_ * 2);
    _Float16* qhT   = (_Float16*)alloc((size_t)B_ * T_ * D_ * 2);
    _Float16* attC  = (_Float16*)alloc((size_t)B_ * C_ * D_ * 2);
    _Float16* attT  = (_Float16*)alloc((size_t)B_ * T_ * D_ * 2);
    _Float16* qwT   = (_Float16*)alloc((size_t)NL_ * 64 * 64 * 2);
    _Float16* kwT   = (_Float16*)alloc((size_t)NL_ * 64 * 128 * 2);
    _Float16* vwT   = (_Float16*)alloc((size_t)NL_ * 64 * 128 * 2);
    _Float16* owT   = (_Float16*)alloc((size_t)NL_ * 64 * 64 * 2);
    _Float16* ewT   = (_Float16*)alloc((size_t)NL_ * 64 * 192 * 2);

    // --- preprocessing ---
    build_order_k<<<B_, 256, 0, stream>>>(MX, MY, X, order, nvalid, mkb,
                                          Tind, Cind, U2, countC, countT);
    make_cmtm_k<<<(B_ * T_ + 255) / 256, 256, 0, stream>>>(countC, countT, CmF, TmF);
    edge_embed_k<<<(int)((BL * D_ + 255) / 256), 256, 0, stream>>>(U2, mkb, edge_w, edge_b, Ubuf[0]);
    tf_init_k<<<(B_ * T_ * D_ + 255) / 256, 256, 0, stream>>>(TX, time_w, time_b, Tfb[0]);
    cf_init_k<<<(B_ * C_ * D_ + 255) / 256, 256, 0, stream>>>(chan_w, chan_b, Cfb[0]);

    // --- weight prep (f32 [K,N] -> f16 [N,K]) ---
    for (int i = 0; i < NL_; ++i) {
        wt_transpose_k<<<(64 * 64 + 255) / 256, 256, 0, stream>>>(qw + i * 64 * 64, qwT + i * 64 * 64, 64, 64);
        wt_transpose_k<<<(128 * 64 + 255) / 256, 256, 0, stream>>>(kw + i * 128 * 64, kwT + i * 64 * 128, 128, 64);
        wt_transpose_k<<<(128 * 64 + 255) / 256, 256, 0, stream>>>(vw + i * 128 * 64, vwT + i * 64 * 128, 128, 64);
        wt_transpose_k<<<(64 * 64 + 255) / 256, 256, 0, stream>>>(ow + i * 64 * 64, owT + i * 64 * 64, 64, 64);
        wt_transpose_k<<<(192 * 64 + 255) / 256, 256, 0, stream>>>(ew + i * 192 * 64, ewT + i * 64 * 192, 192, 64);
    }

    int curb = 0;
    for (int i = 0; i < NL_; ++i) {
        int nxt = 1 - curb;
        // pack current queries to f16
        pack_f16_k<<<(B_ * C_ * D_ + 255) / 256, 256, 0, stream>>>(Cfb[curb], qinC, B_ * C_ * D_);
        pack_f16_k<<<(B_ * T_ * D_ + 255) / 256, 256, 0, stream>>>(Tfb[curb], qinT, B_ * T_ * D_);
        // gather k_t/k_c, build concat inputs
        gather_k<<<(int)BL, 64, 0, stream>>>(Tfb[curb], Cfb[curb], Ubuf[curb],
                                             Tind, Cind, kvinC, kvinT, eg);
        // K/V projections (relu), f16 outputs
        gemm_wmma_k<<<(int)(BL / 16), 128, 0, stream>>>(kvinC, kwT + i * 64 * 128, kb + i * 64,
            nullptr, nullptr, nullptr, khC, (int)BL, 128, 1);
        gemm_wmma_k<<<(int)(BL / 16), 128, 0, stream>>>(kvinC, vwT + i * 64 * 128, vb + i * 64,
            nullptr, nullptr, nullptr, vhC, (int)BL, 128, 1);
        gemm_wmma_k<<<(int)(BL / 16), 128, 0, stream>>>(kvinT, kwT + i * 64 * 128, kb + i * 64,
            nullptr, nullptr, nullptr, khT, (int)BL, 128, 1);
        gemm_wmma_k<<<(int)(BL / 16), 128, 0, stream>>>(kvinT, vwT + i * 64 * 128, vb + i * 64,
            nullptr, nullptr, nullptr, vhT, (int)BL, 128, 1);
        // Q projections (relu)
        gemm_wmma_k<<<B_ * C_ / 16, 128, 0, stream>>>(qinC, qwT + i * 64 * 64, qb + i * 64,
            nullptr, nullptr, nullptr, qhC, B_ * C_, 64, 1);
        gemm_wmma_k<<<B_ * T_ / 16, 128, 0, stream>>>(qinT, qwT + i * 64 * 64, qb + i * 64,
            nullptr, nullptr, nullptr, qhT, B_ * T_, 64, 1);
        // fused flash attention
        attn_k<<<B_ * H_ * (C_ / 16), 32, 0, stream>>>(qhC, khC, vhC, Cind, mkb, attC, C_);
        attn_k<<<B_ * H_ * (T_ / 16), 32, 0, stream>>>(qhT, khT, vhT, Tind, mkb, attT, T_);
        // output projections (+ row-mask), f32 outputs -> next Cf/Tf
        gemm_wmma_k<<<B_ * C_ / 16, 128, 0, stream>>>(attC, owT + i * 64 * 64, ob + i * 64,
            nullptr, CmF, Cfb[nxt], nullptr, B_ * C_, 64, 0);
        gemm_wmma_k<<<B_ * T_ / 16, 128, 0, stream>>>(attT, owT + i * 64 * 64, ob + i * 64,
            nullptr, TmF, Tfb[nxt], nullptr, B_ * T_, 64, 0);
        // edgenn: relu(U + [U|k_t|k_c] @ W + b) * mk -> next U
        gemm_wmma_k<<<(int)(BL / 16), 128, 0, stream>>>(eg, ewT + i * 64 * 192, eb + i * 64,
            Ubuf[curb], mkb, Ubuf[nxt], nullptr, (int)BL, 192, 1);
        curb = nxt;
    }

    // scatter back through the permutation, zero invalid
    scatter_out_k<<<(int)BL, 64, 0, stream>>>(Ubuf[curb], order, nvalid, out);
}
